// SelfAttentionInterview_22273700397197
// MI455X (gfx1250) — matrix operand order repaired and checked by reading, not compiled, not run on hardware
//
#include <hip/hip_runtime.h>
#include <hip/hip_bf16.h>

// MI455X / gfx1250 self-attention (B=4, S=2048, D=1024), f16 WMMA + f32 accum.
// Pipeline: cvt -> proj(QKV; 1/sqrt(D) folded into Q; V stored transposed via
// LDS-staged coalesced writes) -> scores (causal-skipped) -> softmax (writes
// normalized f16 P + zero-filled tail, overlaying the dead Qh/Kh buffers)
// -> P*V (pure f16 WMMA loop).
// No software prefetch in the hot loops: operands are L2-resident (192 MB L2
// holds the whole working set) and the compiler's pipelined clause schedule
// with staggered s_wait_loadcnt gives better overlap than prefetch insertion.

typedef __attribute__((ext_vector_type(16))) _Float16 v16h;
typedef __attribute__((ext_vector_type(8)))  _Float16 v8h;
typedef __attribute__((ext_vector_type(4)))  _Float16 v4h;
typedef __attribute__((ext_vector_type(8)))  float    v8f;

#define BATCH  4
#define SEQ    2048
#define DMODEL 1024
#define MROWS  (BATCH * SEQ)   // 8192

// ---- WMMA helpers ---------------------------------------------------------

__device__ __forceinline__ v8f wmma_f16(v16h a, v16h b, v8f c) {
    // D = A(16x32,f16) * B(32x16,f16) + C(16x16,f32)
    return __builtin_amdgcn_wmma_f32_16x16x32_f16(
        /*neg_a=*/false, a, /*neg_b=*/false, b,
        /*c_mod=*/(short)0, c, /*reuse_a=*/false, /*reuse_b=*/false);
}

// Load 16 halves from two 16B chunks (A- and B-operand: contiguous at +hi*16,
// p1 = p0 + 8).
__device__ __forceinline__ v16h ld2x8h(const _Float16* p0, const _Float16* p1) {
    union { v16h v; v8h h[2]; } u;
    u.h[0] = *(const v8h*)p0;
    u.h[1] = *(const v8h*)p1;
    return u.v;
}

// ---- 0) f32 -> f16 convert -------------------------------------------------

__global__ __launch_bounds__(256)
void cvt_kernel(const float* __restrict__ in, _Float16* __restrict__ out, int n4) {
    int i = blockIdx.x * blockDim.x + threadIdx.x;
    if (i < n4) {
        float4 v = ((const float4*)in)[i];
        v4h h;
        h[0] = (_Float16)v.x; h[1] = (_Float16)v.y;
        h[2] = (_Float16)v.z; h[3] = (_Float16)v.w;
        ((v4h*)out)[i] = h;
    }
}

// ---- 1) QKV projection: C = x * W^T + b -----------------------------------
// Block tile 128x128, 8 waves, each wave 64x32 (4 M-subtiles x 2 N-subtiles).
// which: 0 -> Q (scaled 1/32, row-major), 1 -> K (row-major),
//        2 -> V^T via LDS transpose + coalesced b128 stores.

__global__ __launch_bounds__(256)
void proj_kernel(const _Float16* __restrict__ xh, const _Float16* __restrict__ Wh,
                 const float* __restrict__ bq, const float* __restrict__ bk,
                 const float* __restrict__ bv,
                 _Float16* __restrict__ Qh, _Float16* __restrict__ Kh,
                 _Float16* __restrict__ Vt) {
    __shared__ _Float16 tbuf[128 * 136];   // transpose staging (stride 136: 16B rows, conflict-free)

    const int which = blockIdx.z;
    const _Float16* Wm = Wh + (size_t)which * DMODEL * DMODEL;
    const int m0 = blockIdx.x * 128;
    const int n0 = blockIdx.y * 128;
    const int tid = threadIdx.x;
    const int l = tid & 31, w = tid >> 5;
    const int lo = l & 15, hi = l >> 4;
    const int wm = (w >> 2) * 64;   // 0 or 64
    const int wn = (w & 3) * 32;    // 0,32,64,96

    v8f acc[4][2] = {};
    int arow[4], brow[2];
#pragma unroll
    for (int i = 0; i < 4; ++i) arow[i] = m0 + wm + i * 16 + lo;
#pragma unroll
    for (int j = 0; j < 2; ++j) brow[j] = n0 + wn + j * 16 + lo;

#pragma unroll 2
    for (int k0 = 0; k0 < DMODEL; k0 += 32) {
        v16h b[2];
#pragma unroll
        for (int j = 0; j < 2; ++j) {
            const _Float16* p = Wm + (size_t)brow[j] * DMODEL + k0 + hi * 16;
            b[j] = ld2x8h(p, p + 8);
        }
#pragma unroll
        for (int i = 0; i < 4; ++i) {
            const _Float16* p = xh + (size_t)arow[i] * DMODEL + k0;
            v16h a = ld2x8h(p + hi * 16, p + hi * 16 + 8);
#pragma unroll
            for (int j = 0; j < 2; ++j) acc[i][j] = wmma_f16(a, b[j], acc[i][j]);
        }
    }

    const float* bias = (which == 0) ? bq : (which == 1) ? bk : bv;
    const float scale = (which == 0) ? 0.03125f : 1.0f;  // 1/sqrt(1024)
#pragma unroll
    for (int i = 0; i < 4; ++i) {
#pragma unroll
        for (int j = 0; j < 2; ++j) {
            const int colL = wn + j * 16 + lo;          // local col in [0,128)
            const float bb = bias[n0 + colL];
#pragma unroll
            for (int r = 0; r < 8; ++r) {
                const int rowL = wm + i * 16 + r + 8 * hi;  // local row in [0,128)
                const float v = (acc[i][j][r] + bb) * scale;
                if (which == 0) {
                    Qh[(size_t)(m0 + rowL) * DMODEL + n0 + colL] = (_Float16)v;
                } else if (which == 1) {
                    Kh[(size_t)(m0 + rowL) * DMODEL + n0 + colL] = (_Float16)v;
                } else {
                    tbuf[colL * 136 + rowL] = (_Float16)v;
                }
            }
        }
    }

    if (which == 2) {
        __syncthreads();
        // Coalesced write-out: thread t handles feature line e = t>>1, half s-range.
        const int bt = m0 >> 11;            // batch index
        const int sbase = m0 & (SEQ - 1);   // seq offset within batch
        const int e = tid >> 1;
        const int sh = (tid & 1) * 64;
        _Float16* dst = Vt + ((size_t)(bt * DMODEL + n0 + e)) * SEQ + sbase + sh;
        const _Float16* src = tbuf + e * 136 + sh;
#pragma unroll
        for (int u = 0; u < 8; ++u)
            ((v8h*)dst)[u] = *(const v8h*)(src + u * 8);
    }
}

// ---- 2) scores: S = Qs * K^T (f32), causal tiles only ----------------------

__global__ __launch_bounds__(256)
void scores_kernel(const _Float16* __restrict__ Qh, const _Float16* __restrict__ Kh,
                   float* __restrict__ Sbuf) {
    if (blockIdx.y > blockIdx.x) return;  // tile fully above the diagonal
    const int bat = blockIdx.z;
    const int m0 = blockIdx.x * 128;      // query rows
    const int n0 = blockIdx.y * 128;      // key cols
    const _Float16* Q = Qh + (size_t)bat * SEQ * DMODEL;
    const _Float16* K = Kh + (size_t)bat * SEQ * DMODEL;
    float* Sp = Sbuf + (size_t)bat * SEQ * SEQ;

    const int tid = threadIdx.x;
    const int l = tid & 31, w = tid >> 5;
    const int lo = l & 15, hi = l >> 4;
    const int wm = (w >> 2) * 64, wn = (w & 3) * 32;

    v8f acc[4][2] = {};
    int arow[4], brow[2];
#pragma unroll
    for (int i = 0; i < 4; ++i) arow[i] = m0 + wm + i * 16 + lo;
#pragma unroll
    for (int j = 0; j < 2; ++j) brow[j] = n0 + wn + j * 16 + lo;

#pragma unroll 2
    for (int k0 = 0; k0 < DMODEL; k0 += 32) {
        v16h b[2];
#pragma unroll
        for (int j = 0; j < 2; ++j) {
            const _Float16* p = K + (size_t)brow[j] * DMODEL + k0 + hi * 16;
            b[j] = ld2x8h(p, p + 8);
        }
#pragma unroll
        for (int i = 0; i < 4; ++i) {
            const _Float16* p = Q + (size_t)arow[i] * DMODEL + k0;
            v16h a = ld2x8h(p + hi * 16, p + hi * 16 + 8);
#pragma unroll
            for (int j = 0; j < 2; ++j) acc[i][j] = wmma_f16(a, b[j], acc[i][j]);
        }
    }

#pragma unroll
    for (int i = 0; i < 4; ++i)
#pragma unroll
        for (int j = 0; j < 2; ++j) {
            const int col = n0 + wn + j * 16 + lo;
#pragma unroll
            for (int r = 0; r < 8; ++r) {
                const int row = m0 + wm + i * 16 + r + 8 * hi;
                Sp[(size_t)row * SEQ + col] = acc[i][j][r];
            }
        }
}

// ---- 3) causal softmax: f32 S row -> normalized f16 P row ------------------
// Single global read (row cached in registers, fully unrolled, <=8/thread).
// Zero-fills (q, round_up(q+1,128)) so the PV k-loop reads only valid f16.

__global__ __launch_bounds__(256)
void softmax_kernel(const float* __restrict__ Sbuf, _Float16* __restrict__ Ph) {
    const int q = blockIdx.x, bat = blockIdx.y;
    const float* row = Sbuf + ((size_t)bat * SEQ + q) * SEQ;
    _Float16* prow = Ph + ((size_t)bat * SEQ + q) * SEQ;
    const int t = threadIdx.x;
    const int n = q + 1;                      // valid (unmasked) length
    __shared__ float red[256];

    float e[8];
    float mx = -3.0e38f;
#pragma unroll
    for (int i = 0; i < 8; ++i) {
        const int k = t + i * 256;
        const float v = (k < n) ? row[k] : -3.0e38f;
        e[i] = v;
        mx = fmaxf(mx, v);
    }
    red[t] = mx;
    __syncthreads();
    for (int s = 128; s > 0; s >>= 1) {
        if (t < s) red[t] = fmaxf(red[t], red[t + s]);
        __syncthreads();
    }
    mx = red[0];
    __syncthreads();

    float sum = 0.f;
#pragma unroll
    for (int i = 0; i < 8; ++i) {
        const int k = t + i * 256;
        const float v = (k < n) ? __expf(e[i] - mx) : 0.f;
        e[i] = v;
        sum += v;
    }
    red[t] = sum;
    __syncthreads();
    for (int s = 128; s > 0; s >>= 1) {
        if (t < s) red[t] += red[t + s];
        __syncthreads();
    }
    const float inv = 1.f / red[0];

#pragma unroll
    for (int i = 0; i < 8; ++i) {
        const int k = t + i * 256;
        if (k < n) prow[k] = (_Float16)(e[i] * inv);
    }
    const int zend = ((q + 128) >> 7) << 7;   // round_up(q+1, 128); zend-n < 256
    const int k = n + t;
    if (k < zend) prow[k] = (_Float16)0.f;
}

// ---- 4) O = P * V  (both operands f16; B = V^T is K-contiguous) ------------

__global__ __launch_bounds__(256)
void pv_kernel(const _Float16* __restrict__ Ph, const _Float16* __restrict__ Vt,
               float* __restrict__ out) {
    const int bat = blockIdx.z;
    const int m0 = blockIdx.x * 128;   // query rows
    const int n0 = blockIdx.y * 128;   // feature cols
    const _Float16* P = Ph + (size_t)bat * SEQ * SEQ;
    const _Float16* V = Vt + (size_t)bat * DMODEL * SEQ;
    float* O = out + (size_t)bat * SEQ * DMODEL;

    const int tid = threadIdx.x;
    const int l = tid & 31, w = tid >> 5;
    const int lo = l & 15, hi = l >> 4;
    const int wm = (w >> 2) * 64, wn = (w & 3) * 32;

    v8f acc[4][2] = {};
    int arow[4], brow[2];
#pragma unroll
    for (int i = 0; i < 4; ++i) arow[i] = m0 + wm + i * 16 + lo;
#pragma unroll
    for (int j = 0; j < 2; ++j) brow[j] = n0 + wn + j * 16 + lo;

    const int kend = m0 + 128;  // causal bound (P zero-filled up to here)
#pragma unroll 2
    for (int k0 = 0; k0 < kend; k0 += 32) {
        v16h b[2];
#pragma unroll
        for (int j = 0; j < 2; ++j) {
            const _Float16* p = V + (size_t)brow[j] * SEQ + k0 + hi * 16;
            b[j] = ld2x8h(p, p + 8);
        }
#pragma unroll
        for (int i = 0; i < 4; ++i) {
            const _Float16* p = P + (size_t)arow[i] * SEQ + k0;
            v16h a = ld2x8h(p + hi * 16, p + hi * 16 + 8);
#pragma unroll
            for (int j = 0; j < 2; ++j) acc[i][j] = wmma_f16(a, b[j], acc[i][j]);
        }
    }

#pragma unroll
    for (int i = 0; i < 4; ++i)
#pragma unroll
        for (int j = 0; j < 2; ++j) {
            const int col = n0 + wn + j * 16 + lo;
#pragma unroll
            for (int r = 0; r < 8; ++r) {
                const int row = m0 + wm + i * 16 + r + 8 * hi;
                O[(size_t)row * DMODEL + col] = acc[i][j][r];
            }
        }
}

// ---- launcher --------------------------------------------------------------

extern "C" void kernel_launch(void* const* d_in, const int* in_sizes, int n_in,
                              void* d_out, int out_size, void* d_ws, size_t ws_size,
                              hipStream_t stream) {
    (void)in_sizes; (void)n_in; (void)out_size; (void)ws_size;

    const float* x  = (const float*)d_in[0];
    // d_in[1] = causal mask: implicit, unused
    const float* Wq = (const float*)d_in[2];
    const float* bq = (const float*)d_in[3];
    const float* Wk = (const float*)d_in[4];
    const float* bk = (const float*)d_in[5];
    const float* Wv = (const float*)d_in[6];
    const float* bv = (const float*)d_in[7];
    float* out = (float*)d_out;

    char* ws = (char*)d_ws;
    size_t off = 0;
    _Float16* xh = (_Float16*)(ws + off); off += (size_t)MROWS * DMODEL * 2;      // 16 MB
    _Float16* Wh = (_Float16*)(ws + off); off += (size_t)3 * DMODEL * DMODEL * 2; // 6 MB
    _Float16* Qh = (_Float16*)(ws + off); off += (size_t)MROWS * DMODEL * 2;      // 16 MB
    _Float16* Kh = (_Float16*)(ws + off); off += (size_t)MROWS * DMODEL * 2;      // 16 MB
    _Float16* Vt = (_Float16*)(ws + off); off += (size_t)MROWS * DMODEL * 2;      // 16 MB
    float*  Sbuf = (float*)(ws + off);    off += (size_t)BATCH * SEQ * SEQ * 4;   // 64 MB
    // P (f16, 32 MB) overlays Qh+Kh (dead after scores_kernel; exactly 32 MB).
    _Float16* Phf = Qh;

    const int n4x = MROWS * DMODEL / 4;
    cvt_kernel<<<(n4x + 255) / 256, 256, 0, stream>>>(x, xh, n4x);
    const int n4w = DMODEL * DMODEL / 4;
    cvt_kernel<<<(n4w + 255) / 256, 256, 0, stream>>>(Wq, Wh, n4w);
    cvt_kernel<<<(n4w + 255) / 256, 256, 0, stream>>>(Wk, Wh + (size_t)DMODEL * DMODEL, n4w);
    cvt_kernel<<<(n4w + 255) / 256, 256, 0, stream>>>(Wv, Wh + (size_t)2 * DMODEL * DMODEL, n4w);

    proj_kernel<<<dim3(MROWS / 128, DMODEL / 128, 3), 256, 0, stream>>>(
        xh, Wh, bq, bk, bv, Qh, Kh, Vt);
    scores_kernel<<<dim3(SEQ / 128, SEQ / 128, BATCH), 256, 0, stream>>>(Qh, Kh, Sbuf);
    softmax_kernel<<<dim3(SEQ, BATCH), 256, 0, stream>>>(Sbuf, Phf);
    pv_kernel<<<dim3(SEQ / 128, DMODEL / 128, BATCH), 256, 0, stream>>>(Phf, Vt, out);
}
